// CleanGeneralDIT_17154099380432
// MI455X (gfx1250) — matrix operand order repaired and checked by reading, not compiled, not run on hardware
//
#include <hip/hip_runtime.h>
#include <hip/hip_bf16.h>
#include <math.h>

typedef __attribute__((ext_vector_type(16))) _Float16 v16h;
typedef __attribute__((ext_vector_type(8)))  _Float16 v8h;
typedef __attribute__((ext_vector_type(8)))  float    v8f;
typedef __attribute__((ext_vector_type(4)))  unsigned u32x4;
typedef __attribute__((ext_vector_type(8)))  unsigned u32x8;

#define LSEQ   3072
#define DMODEL 1024
#define NHEAD  16
#define HDIM   64
#define SCTX   512
#define NBLK   2
#define DFFN   4096

// ---------------------------------------------------------------------------
// Tensor Data Mover: async 2D tile (rows x row_bytes) global -> LDS.
// D# built in SGPRs; 2-group form (2D tensor). data_size = 8 bytes.
// Tracked by TENSORcnt; caller must s_wait_tensorcnt before consuming.
// ---------------------------------------------------------------------------
__device__ __forceinline__ unsigned lds_addr32(const void* p) {
  return (unsigned)(size_t)p;   // generic LDS pointer: low 32 bits = LDS offset
}

__device__ __forceinline__ void tdm_load_2d(unsigned lds_addr, const void* gptr,
                                            unsigned rows, unsigned row_bytes,
                                            unsigned stride_bytes) {
  const unsigned tile0    = row_bytes >> 3;     // 8-byte units
  const unsigned stride_u = stride_bytes >> 3;
  const unsigned long long ga = (unsigned long long)gptr;
  u32x4 g0;
  g0[0] = 1u;                                             // count=1 (valid), user mode
  g0[1] = lds_addr;                                       // lds_addr (bytes)
  g0[2] = (unsigned)ga;                                   // global_addr[31:0]
  g0[3] = ((unsigned)(ga >> 32) & 0x01FFFFFFu) | (2u << 30);  // addr[56:32] | type=2
  u32x8 g1;
  g1[0] = 3u << 16;                       // workgroup_mask=0, data_size=3 (8B)
  g1[1] = (tile0 & 0xFFFFu) << 16;        // tensor_dim0[15:0]
  g1[2] = (rows  & 0xFFFFu) << 16;        // tensor_dim0[31:16]=0 | tensor_dim1[15:0]
  g1[3] = (tile0 & 0xFFFFu) << 16;        // tensor_dim1[31:16]=0 | tile_dim0
  g1[4] = rows & 0xFFFFu;                 // tile_dim1 | tile_dim2=0
  g1[5] = stride_u;                       // tensor_dim0_stride[31:0]
  g1[6] = 0u;                             // stride[47:32]=0 | dim1_stride[15:0]=0
  g1[7] = 0u;                             // dim1_stride[47:16]=0
  asm volatile("tensor_load_to_lds %0, %1" :: "s"(g0), "s"(g1) : "memory");
}

// ---------------------------------------------------------------------------
// fp32 -> fp16 conversion (weights / context staging)
// ---------------------------------------------------------------------------
__global__ void cvt_f32_f16(const float* __restrict__ src, _Float16* __restrict__ dst, int n) {
  int i = (blockIdx.x * blockDim.x + threadIdx.x) * 4;
  if (i + 3 < n) {
    float4 v = *(const float4*)(src + i);
    dst[i + 0] = (_Float16)v.x;
    dst[i + 1] = (_Float16)v.y;
    dst[i + 2] = (_Float16)v.z;
    dst[i + 3] = (_Float16)v.w;
  } else {
    for (int j = i; j < n; ++j) dst[j] = (_Float16)src[j];
  }
}

// ---------------------------------------------------------------------------
// AdaLN modulation GEMV: mods = temb @ adaW_i^T + adab_i  (1 x 6144)
// ---------------------------------------------------------------------------
__global__ void ada_gemv(const float* __restrict__ t, const float* __restrict__ W,
                         const float* __restrict__ b, float* __restrict__ out,
                         int nout, int K) {
  int j = blockIdx.x * blockDim.x + threadIdx.x;
  if (j >= nout) return;
  const float* w = W + (size_t)j * K;
  float s = 0.f;
  for (int k = 0; k < K; ++k) s += t[k] * w[k];
  out[j] = s + b[j];
}

// ---------------------------------------------------------------------------
// LayerNorm (+ optional scale/shift modulation), fp16 output. One row / block.
// ---------------------------------------------------------------------------
__global__ void ln_mod(const float* __restrict__ X, const float* __restrict__ sc,
                       const float* __restrict__ sh, _Float16* __restrict__ out) {
  __shared__ float red0[8];
  __shared__ float red1[8];
  const int row = blockIdx.x;
  const float* xr = X + (size_t)row * DMODEL;
  float xs[4];
  float s = 0.f;
#pragma unroll
  for (int i = 0; i < 4; ++i) { xs[i] = xr[threadIdx.x + i * 256]; s += xs[i]; }
#pragma unroll
  for (int m = 16; m; m >>= 1) s += __shfl_xor(s, m, 32);
  const int wave = threadIdx.x >> 5, lane = threadIdx.x & 31;
  if (lane == 0) red0[wave] = s;
  __syncthreads();
  float mean = 0.f;
#pragma unroll
  for (int w = 0; w < 8; ++w) mean += red0[w];
  mean *= (1.0f / DMODEL);
  float v = 0.f;
#pragma unroll
  for (int i = 0; i < 4; ++i) { float d = xs[i] - mean; v += d * d; }
#pragma unroll
  for (int m = 16; m; m >>= 1) v += __shfl_xor(v, m, 32);
  if (lane == 0) red1[wave] = v;
  __syncthreads();
  float var = 0.f;
#pragma unroll
  for (int w = 0; w < 8; ++w) var += red1[w];
  var *= (1.0f / DMODEL);
  const float rsq = rsqrtf(var + 1e-6f);
#pragma unroll
  for (int i = 0; i < 4; ++i) {
    const int c = threadIdx.x + i * 256;
    float y = (xs[i] - mean) * rsq;
    if (sc) y = y * (1.f + sc[c]) + sh[c];
    out[(size_t)row * DMODEL + c] = (_Float16)y;
  }
}

// ---------------------------------------------------------------------------
// RoPE (rot-half form) + scale + fp16 convert.  X,out: [L, D]; rope: [L, 64]
// ---------------------------------------------------------------------------
__global__ void rope_scale_cvt(const float* __restrict__ X, const float* __restrict__ rope,
                               _Float16* __restrict__ out, float alpha) {
  const int idx = blockIdx.x * blockDim.x + threadIdx.x;   // over L*D
  const int l = idx >> 10;
  const int c = idx & 1023;
  const int e = c & 63;
  const float r  = rope[l * 64 + e];
  const float cs = cosf(r);
  const float sn = sinf(r);
  const int   partner = (e < 32) ? c + 32 : c - 32;
  const float sgn     = (e < 32) ? -1.f : 1.f;
  const size_t base = (size_t)l * DMODEL;
  const float v = X[base + c] * cs + sgn * X[base + partner] * sn;
  out[base + c] = (_Float16)(v * alpha);
}

// ---------------------------------------------------------------------------
// WMMA GEMM: C[M,N] = A[M,K] (f16, row-major) * W[N,K]^T (f16, row-major)
// 128x128 tile / 256-thread block (8 waves, 2x4), K-step 32.
// TDM double-buffered LDS staging (wave 0 issues, TENSORcnt pipelined).
// MODE 0: store f32            MODE 1: store f16 * alpha
// MODE 2: +bias, exact GELU, store f16
// MODE 3: C32[m,n] += gate[n]*(v + bias[n])   (gate/bias nullable)
// ---------------------------------------------------------------------------
template <int MODE>
__global__ void gemm_wmma(const _Float16* __restrict__ A, const _Float16* __restrict__ W,
                          float* __restrict__ C32, _Float16* __restrict__ C16,
                          const float* __restrict__ bias, const float* __restrict__ gate,
                          int M, int N, int K, float alpha) {
  __shared__ __attribute__((aligned(16))) _Float16 As[2][128 * 32];
  __shared__ __attribute__((aligned(16))) _Float16 Bs[2][128 * 32];
  const int tid  = threadIdx.x;
  const int wave = tid >> 5, lane = tid & 31;
  const int lh = lane & 15, ls = lane >> 4;
  const int wm = wave >> 2, wn = wave & 3;   // 2 x 4 wave grid
  const int m0 = blockIdx.y * 128, n0 = blockIdx.x * 128;

  v8f acc[4][2];
  v8f vzero = {};
#pragma unroll
  for (int a = 0; a < 4; ++a)
#pragma unroll
    for (int b = 0; b < 2; ++b) acc[a][b] = vzero;

  // scalar (SGPR) condition so exactly one wave issues TDM ops
  const bool issuer = ((__builtin_amdgcn_readfirstlane((int)threadIdx.x) >> 5) == 0);
  const _Float16* Abase = A + (size_t)m0 * K;
  const _Float16* Wbase = W + (size_t)n0 * K;

  if (issuer) {   // prologue: stage k-tile 0 into buffer 0
    tdm_load_2d(lds_addr32(&As[0][0]), Abase, 128, 64, K * 2);
    tdm_load_2d(lds_addr32(&Bs[0][0]), Wbase, 128, 64, K * 2);
  }

  int cur = 0;
  for (int k0 = 0; k0 < K; k0 += 32, cur ^= 1) {
    __syncthreads();   // everyone done reading the buffer we are about to refill
    if (issuer) {
      if (k0 + 32 < K) {
        tdm_load_2d(lds_addr32(&As[cur ^ 1][0]), Abase + k0 + 32, 128, 64, K * 2);
        tdm_load_2d(lds_addr32(&Bs[cur ^ 1][0]), Wbase + k0 + 32, 128, 64, K * 2);
        __builtin_amdgcn_s_wait_tensorcnt((short)2);  // current pair has landed
      } else {
        __builtin_amdgcn_s_wait_tensorcnt((short)0);
      }
    }
    __syncthreads();   // current buffer visible to all waves

    v16h af[4], bf[2];
#pragma unroll
    for (int fr = 0; fr < 4; ++fr) {   // A frag: lane row, K lo/hi per lane half
      const _Float16* p = &As[cur][(wm * 64 + fr * 16 + lh) * 32 + ls * 8];
      v8h lo = *(const v8h*)p;
      v8h hi = *(const v8h*)(p + 16);
#pragma unroll
      for (int i = 0; i < 8; ++i) { af[fr][i] = lo[i]; af[fr][i + 8] = hi[i]; }
    }
#pragma unroll
    for (int fc = 0; fc < 2; ++fc) {   // B frag: lane = output col, K lo/hi per lane half
      const _Float16* p = &Bs[cur][(wn * 32 + fc * 16 + lh) * 32 + ls * 8];
      v8h lo = *(const v8h*)p;
      v8h hi = *(const v8h*)(p + 16);
#pragma unroll
      for (int i = 0; i < 8; ++i) { bf[fc][i] = lo[i]; bf[fc][i + 8] = hi[i]; }
    }
#pragma unroll
    for (int fr = 0; fr < 4; ++fr)
#pragma unroll
      for (int fc = 0; fc < 2; ++fc)
        acc[fr][fc] = __builtin_amdgcn_wmma_f32_16x16x32_f16(
            false, af[fr], false, bf[fc], (short)0, acc[fr][fc], false, false);
  }

#pragma unroll
  for (int fr = 0; fr < 4; ++fr)
#pragma unroll
    for (int fc = 0; fc < 2; ++fc) {
      const int nn = n0 + wn * 32 + fc * 16 + lh;
#pragma unroll
      for (int r = 0; r < 8; ++r) {
        const int mm = m0 + wm * 64 + fr * 16 + r + ls * 8;
        float v = acc[fr][fc][r];
        const size_t o = (size_t)mm * N + nn;
        if (MODE == 0) {
          C32[o] = v;
        } else if (MODE == 1) {
          C16[o] = (_Float16)(v * alpha);
        } else if (MODE == 2) {
          v += bias[nn];
          v = 0.5f * v * (1.f + erff(v * 0.70710678f));
          C16[o] = (_Float16)v;
        } else {
          const float g  = gate ? gate[nn] : 1.f;
          const float bb = bias ? bias[nn] : 0.f;
          C32[o] += g * (v + bb);
        }
      }
    }
}

// ---------------------------------------------------------------------------
// Flash-style attention, WMMA for QK^T and PV. One head x 64 query rows per
// 128-thread block (4 waves, 16 rows each). Q pre-scaled by hd^-0.5.
// K chunk staged by TDM; V chunk transposed into LDS by the waves.
// ---------------------------------------------------------------------------
__global__ void attn_wmma(const _Float16* __restrict__ Q, const _Float16* __restrict__ Kh,
                          const _Float16* __restrict__ Vh, _Float16* __restrict__ O,
                          int Lk) {
  __shared__ __attribute__((aligned(16))) _Float16 Ks[32 * 64];   // [key][d]
  __shared__ __attribute__((aligned(16))) _Float16 Vt[64 * 32];   // [d][key]
  __shared__ __attribute__((aligned(16))) _Float16 Ps[4 * 16 * 32];
  const int tid  = threadIdx.x;
  const int wave = tid >> 5, lane = tid & 31;
  const int lh = lane & 15, ls = lane >> 4;
  const int h    = blockIdx.y;
  const int q0   = blockIdx.x * 64;
  const int col0 = h * HDIM;
  const bool issuer = ((__builtin_amdgcn_readfirstlane((int)threadIdx.x) >> 5) == 0);

  // Q fragments (A-layout) straight from global: row-per-lane, d lo/hi per lane half.
  v16h qf[2];
  {
    const _Float16* qp = Q + (size_t)(q0 + wave * 16 + lh) * DMODEL + col0;
#pragma unroll
    for (int dh = 0; dh < 2; ++dh) {
      v8h lo = *(const v8h*)(qp + dh * 32 + ls * 8);
      v8h hi = *(const v8h*)(qp + dh * 32 + ls * 8 + 16);
#pragma unroll
      for (int i = 0; i < 8; ++i) { qf[dh][i] = lo[i]; qf[dh][i + 8] = hi[i]; }
    }
  }

  v8f oacc[4];
  v8f vzero = {};
#pragma unroll
  for (int dt = 0; dt < 4; ++dt) oacc[dt] = vzero;
  float mrow[8], lrow[8];
#pragma unroll
  for (int r = 0; r < 8; ++r) { mrow[r] = -3.0e38f; lrow[r] = 0.f; }

  for (int kc = 0; kc < Lk; kc += 32) {
    __syncthreads();
    if (issuer)   // stage K chunk [32 x 64] via Tensor Data Mover
      tdm_load_2d(lds_addr32(&Ks[0]), Kh + (size_t)kc * DMODEL + col0,
                  32, 128, DMODEL * 2);
    for (int idx = tid; idx < 2048; idx += 128) {  // stage V transposed [64 x 32]
      const int key = idx & 31, dd = idx >> 5;
      Vt[dd * 32 + key] = Vh[(size_t)(kc + key) * DMODEL + col0 + dd];
    }
    if (issuer) __builtin_amdgcn_s_wait_tensorcnt((short)0);
    __syncthreads();

    // S = Q * K^T  (two 16-key tiles, K-dim = 64 in two WMMA steps)
    v8f s0 = vzero, s1 = vzero;
#pragma unroll
    for (int dh = 0; dh < 2; ++dh) {
      const _Float16* p0 = &Ks[(0 * 16 + lh) * 64 + dh * 32 + ls * 8];
      const _Float16* p1 = &Ks[(1 * 16 + lh) * 64 + dh * 32 + ls * 8];
      v8h a0 = *(const v8h*)p0, a1 = *(const v8h*)(p0 + 16);
      v8h c0 = *(const v8h*)p1, c1 = *(const v8h*)(p1 + 16);
      v16h b0, b1;
#pragma unroll
      for (int i = 0; i < 8; ++i) {
        b0[i] = a0[i]; b0[i + 8] = a1[i];
        b1[i] = c0[i]; b1[i + 8] = c1[i];
      }
      s0 = __builtin_amdgcn_wmma_f32_16x16x32_f16(false, qf[dh], false, b0, (short)0, s0, false, false);
      s1 = __builtin_amdgcn_wmma_f32_16x16x32_f16(false, qf[dh], false, b1, (short)0, s1, false, false);
    }

    // online softmax (row r lives in VGPR r, lane halves = row +/- 8)
    float p0v[8], p1v[8];
#pragma unroll
    for (int r = 0; r < 8; ++r) {
      float rmax = fmaxf(s0[r], s1[r]);
#pragma unroll
      for (int m = 8; m; m >>= 1) rmax = fmaxf(rmax, __shfl_xor(rmax, m, 16));
      const float mn   = fmaxf(mrow[r], rmax);
      const float corr = __expf(mrow[r] - mn);
      const float e0   = __expf(s0[r] - mn);
      const float e1   = __expf(s1[r] - mn);
      float rs = e0 + e1;
#pragma unroll
      for (int m = 8; m; m >>= 1) rs += __shfl_xor(rs, m, 16);
      lrow[r] = lrow[r] * corr + rs;
      mrow[r] = mn;
      p0v[r] = e0; p1v[r] = e1;
#pragma unroll
      for (int dt = 0; dt < 4; ++dt) oacc[dt][r] *= corr;
    }

    // C-layout -> A-layout for P via LDS
#pragma unroll
    for (int r = 0; r < 8; ++r) {
      const int mr = r + ls * 8;
      Ps[wave * 512 + mr * 32 + 0 * 16 + lh] = (_Float16)p0v[r];
      Ps[wave * 512 + mr * 32 + 1 * 16 + lh] = (_Float16)p1v[r];
    }
    __syncthreads();

    v16h pf;
    {
      const _Float16* pp = &Ps[wave * 512 + lh * 32 + ls * 8];
      v8h lo = *(const v8h*)pp;
      v8h hi = *(const v8h*)(pp + 16);
#pragma unroll
      for (int i = 0; i < 8; ++i) { pf[i] = lo[i]; pf[i + 8] = hi[i]; }
    }
#pragma unroll
    for (int dt = 0; dt < 4; ++dt) {   // O += P * V
      const _Float16* vp = &Vt[(dt * 16 + lh) * 32 + ls * 8];
      v8h lo = *(const v8h*)vp, hi = *(const v8h*)(vp + 16);
      v16h vf;
#pragma unroll
      for (int i = 0; i < 8; ++i) { vf[i] = lo[i]; vf[i + 8] = hi[i]; }
      oacc[dt] = __builtin_amdgcn_wmma_f32_16x16x32_f16(false, pf, false, vf, (short)0, oacc[dt], false, false);
    }
  }

#pragma unroll
  for (int dt = 0; dt < 4; ++dt)
#pragma unroll
    for (int r = 0; r < 8; ++r) {
      const int row = q0 + wave * 16 + r + ls * 8;
      const float ov = oacc[dt][r] / lrow[r];
      O[(size_t)row * DMODEL + col0 + dt * 16 + lh] = (_Float16)ov;
    }
}

// ---------------------------------------------------------------------------
// Host orchestration
// ---------------------------------------------------------------------------
extern "C" void kernel_launch(void* const* d_in, const int* in_sizes, int n_in,
                              void* d_out, int out_size, void* d_ws, size_t ws_size,
                              hipStream_t stream) {
  (void)in_sizes; (void)n_in; (void)out_size; (void)ws_size;
  const float* x_in = (const float*)d_in[0];
  const float* temb = (const float*)d_in[1];
  const float* ctx  = (const float*)d_in[2];
  const float* rope = (const float*)d_in[3];
  const float* adaW = (const float*)d_in[4];
  const float* adab = (const float*)d_in[5];
  const float* sqW  = (const float*)d_in[6];
  const float* skW  = (const float*)d_in[7];
  const float* svW  = (const float*)d_in[8];
  const float* soW  = (const float*)d_in[9];
  const float* sob  = (const float*)d_in[10];
  const float* cqW  = (const float*)d_in[11];
  const float* ckW  = (const float*)d_in[12];
  const float* cvW  = (const float*)d_in[13];
  const float* coW  = (const float*)d_in[14];
  const float* cob  = (const float*)d_in[15];
  const float* f1W  = (const float*)d_in[16];
  const float* f1b  = (const float*)d_in[17];
  const float* f2W  = (const float*)d_in[18];
  const float* f2b  = (const float*)d_in[19];
  float* X = (float*)d_out;   // running residual stream

  // ---- workspace carve-out -------------------------------------------------
  char* p = (char*)d_ws;
  auto take = [&](size_t bytes) -> char* {
    char* r = p;
    p += (bytes + 255) & ~(size_t)255;
    return r;
  };
  const size_t WDD = (size_t)NBLK * DMODEL * DMODEL;
  const size_t WFD = (size_t)NBLK * DFFN * DMODEL;
  const size_t LD  = (size_t)LSEQ * DMODEL;
  const size_t SD  = (size_t)SCTX * DMODEL;

  _Float16* sqWh = (_Float16*)take(WDD * 2);
  _Float16* skWh = (_Float16*)take(WDD * 2);
  _Float16* svWh = (_Float16*)take(WDD * 2);
  _Float16* soWh = (_Float16*)take(WDD * 2);
  _Float16* cqWh = (_Float16*)take(WDD * 2);
  _Float16* ckWh = (_Float16*)take(WDD * 2);
  _Float16* cvWh = (_Float16*)take(WDD * 2);
  _Float16* coWh = (_Float16*)take(WDD * 2);
  _Float16* f1Wh = (_Float16*)take(WFD * 2);
  _Float16* f2Wh = (_Float16*)take(WFD * 2);
  _Float16* ctxh = (_Float16*)take(SD * 2);
  float*    mods = (float*)take((size_t)NBLK * 6 * DMODEL * 4);
  _Float16* hxh  = (_Float16*)take(LD * 2);
  float*    qf32 = (float*)take(LD * 4);
  float*    kf32 = (float*)take(LD * 4);
  _Float16* qh   = (_Float16*)take(LD * 2);
  _Float16* kh   = (_Float16*)take(LD * 2);
  _Float16* vh   = (_Float16*)take(LD * 2);
  _Float16* ckh  = (_Float16*)take(SD * 2);
  _Float16* cvh  = (_Float16*)take(SD * 2);
  _Float16* oh   = (_Float16*)take(LD * 2);
  _Float16* h1h  = (_Float16*)take((size_t)LSEQ * DFFN * 2);

  auto cvt = [&](const float* s, _Float16* d, size_t n) {
    cvt_f32_f16<<<dim3((unsigned)((n + 1023) / 1024)), dim3(256), 0, stream>>>(s, d, (int)n);
  };
  auto gemm = [&](int mode, const _Float16* A, const _Float16* Wt, float* c32, _Float16* c16,
                  const float* bias, const float* gate, int M, int N, int K, float alpha) {
    dim3 g(N / 128, M / 128), b(256);
    switch (mode) {
      case 0: gemm_wmma<0><<<g, b, 0, stream>>>(A, Wt, c32, c16, bias, gate, M, N, K, alpha); break;
      case 1: gemm_wmma<1><<<g, b, 0, stream>>>(A, Wt, c32, c16, bias, gate, M, N, K, alpha); break;
      case 2: gemm_wmma<2><<<g, b, 0, stream>>>(A, Wt, c32, c16, bias, gate, M, N, K, alpha); break;
      default: gemm_wmma<3><<<g, b, 0, stream>>>(A, Wt, c32, c16, bias, gate, M, N, K, alpha); break;
    }
  };

  // seed residual stream: X = x
  hipMemcpyAsync(X, x_in, LD * sizeof(float), hipMemcpyDeviceToDevice, stream);

  // one-time f16 staging of all weights + context
  cvt(sqW, sqWh, WDD); cvt(skW, skWh, WDD); cvt(svW, svWh, WDD); cvt(soW, soWh, WDD);
  cvt(cqW, cqWh, WDD); cvt(ckW, ckWh, WDD); cvt(cvW, cvWh, WDD); cvt(coW, coWh, WDD);
  cvt(f1W, f1Wh, WFD); cvt(f2W, f2Wh, WFD);
  cvt(ctx, ctxh, SD);

  const float scale = 0.125f;  // hd^-0.5 = 64^-0.5

  for (int i = 0; i < NBLK; ++i) {
    float* mb = mods + (size_t)i * 6 * DMODEL;
    const float* sh_msa = mb + 0 * DMODEL;
    const float* sc_msa = mb + 1 * DMODEL;
    const float* g_msa  = mb + 2 * DMODEL;
    const float* sh_mlp = mb + 3 * DMODEL;
    const float* sc_mlp = mb + 4 * DMODEL;
    const float* g_mlp  = mb + 5 * DMODEL;

    // AdaLN modulation GEMV
    ada_gemv<<<dim3(24), dim3(256), 0, stream>>>(
        temb, adaW + (size_t)i * 6 * DMODEL * DMODEL, adab + (size_t)i * 6 * DMODEL,
        mb, 6 * DMODEL, DMODEL);

    // ---- self attention -----------------------------------------------------
    ln_mod<<<dim3(LSEQ), dim3(256), 0, stream>>>(X, sc_msa, sh_msa, hxh);
    gemm(0, hxh, sqWh + (size_t)i * DMODEL * DMODEL, qf32, nullptr, nullptr, nullptr,
         LSEQ, DMODEL, DMODEL, 1.f);
    gemm(0, hxh, skWh + (size_t)i * DMODEL * DMODEL, kf32, nullptr, nullptr, nullptr,
         LSEQ, DMODEL, DMODEL, 1.f);
    gemm(1, hxh, svWh + (size_t)i * DMODEL * DMODEL, nullptr, vh, nullptr, nullptr,
         LSEQ, DMODEL, DMODEL, 1.f);
    rope_scale_cvt<<<dim3(LSEQ * DMODEL / 256), dim3(256), 0, stream>>>(qf32, rope, qh, scale);
    rope_scale_cvt<<<dim3(LSEQ * DMODEL / 256), dim3(256), 0, stream>>>(kf32, rope, kh, 1.f);
    attn_wmma<<<dim3(LSEQ / 64, NHEAD), dim3(128), 0, stream>>>(qh, kh, vh, oh, LSEQ);
    gemm(3, oh, soWh + (size_t)i * DMODEL * DMODEL, X, nullptr,
         sob + (size_t)i * DMODEL, g_msa, LSEQ, DMODEL, DMODEL, 1.f);

    // ---- cross attention ----------------------------------------------------
    ln_mod<<<dim3(LSEQ), dim3(256), 0, stream>>>(X, nullptr, nullptr, hxh);
    gemm(1, hxh, cqWh + (size_t)i * DMODEL * DMODEL, nullptr, qh, nullptr, nullptr,
         LSEQ, DMODEL, DMODEL, scale);
    gemm(1, ctxh, ckWh + (size_t)i * DMODEL * DMODEL, nullptr, ckh, nullptr, nullptr,
         SCTX, DMODEL, DMODEL, 1.f);
    gemm(1, ctxh, cvWh + (size_t)i * DMODEL * DMODEL, nullptr, cvh, nullptr, nullptr,
         SCTX, DMODEL, DMODEL, 1.f);
    attn_wmma<<<dim3(LSEQ / 64, NHEAD), dim3(128), 0, stream>>>(qh, ckh, cvh, oh, SCTX);
    gemm(3, oh, coWh + (size_t)i * DMODEL * DMODEL, X, nullptr,
         cob + (size_t)i * DMODEL, nullptr, LSEQ, DMODEL, DMODEL, 1.f);

    // ---- MLP ----------------------------------------------------------------
    ln_mod<<<dim3(LSEQ), dim3(256), 0, stream>>>(X, sc_mlp, sh_mlp, hxh);
    gemm(2, hxh, f1Wh + (size_t)i * DFFN * DMODEL, nullptr, h1h,
         f1b + (size_t)i * DFFN, nullptr, LSEQ, DFFN, DMODEL, 1.f);
    gemm(3, h1h, f2Wh + (size_t)i * DMODEL * DFFN, X, nullptr,
         f2b + (size_t)i * DMODEL, g_mlp, LSEQ, DMODEL, DFFN, 1.f);
  }
}
